// sLSTMLayer_77395310674067
// MI455X (gfx1250) — matrix operand order repaired and checked by reading, not compiled, not run on hardware
//
#include <hip/hip_runtime.h>
#include <hip/hip_bf16.h>
#include <stdint.h>

#define BDIM  8
#define SDIM  2048
#define EDIM  1024
#define NHEAD 4
#define DHEAD 256
#define KCONV 4
#define NGATE 4

typedef __attribute__((ext_vector_type(16))) __bf16 v16bf;
typedef __attribute__((ext_vector_type(8)))  float  v8f;
typedef __attribute__((ext_vector_type(4)))  int    v4i;

static __device__ __forceinline__ unsigned short f2bf(float f) {
  unsigned int u = __float_as_uint(f);
  u += 0x7FFFu + ((u >> 16) & 1u);           // round-to-nearest-even truncation
  return (unsigned short)(u >> 16);
}

union FragCast {
  struct { uint4 lo, hi; } q;                 // 32 bytes
  v16bf v;                                    // 16 x bf16 = 32 bytes
};

// ---------------------------------------------------------------------------
// CDNA5 async global->LDS copy (ASYNCcnt-tracked), with asm fallback.
// Builtin signature (probe-confirmed param 1): (v4i AS(1)*, v4i AS(3)*, i32, i32)
// ---------------------------------------------------------------------------
static __device__ __forceinline__ void async_copy_b128(const void* gsrc, void* ldst) {
#if __has_builtin(__builtin_amdgcn_global_load_async_to_lds_b128)
  __builtin_amdgcn_global_load_async_to_lds_b128(
      (__attribute__((address_space(1))) v4i*)gsrc,
      (__attribute__((address_space(3))) v4i*)ldst, 0, 0);
#else
  unsigned int loff = (unsigned int)(unsigned long long)(uintptr_t)ldst;
  unsigned long long ga = (unsigned long long)(uintptr_t)gsrc;
  asm volatile("global_load_async_to_lds_b128 %0, %1, off"
               :: "v"(loff), "v"(ga) : "memory");
#endif
}

static __device__ __forceinline__ void wait_async0() {
#if __has_builtin(__builtin_amdgcn_s_wait_asynccnt)
  __builtin_amdgcn_s_wait_asynccnt(0);
#else
  asm volatile("s_wait_asynccnt 0x0" ::: "memory");
#endif
}

static __device__ __forceinline__ void wait_async2() {
#if __has_builtin(__builtin_amdgcn_s_wait_asynccnt)
  __builtin_amdgcn_s_wait_asynccnt(2);
#else
  asm volatile("s_wait_asynccnt 0x2" ::: "memory");
#endif
}

// ---------------------------------------------------------------------------
// Kernel 0: weight prep — bf16 gate weights (kept (h,o,d) row-major so WMMA B
// fragments read K(d)-contiguous) and rkernel transposed to (h, j, d) bf16 so
// scan B fragments also read K(d)-contiguous.
// ---------------------------------------------------------------------------
__global__ __launch_bounds__(256) void prep_kernel(
    const float* __restrict__ fg, const float* __restrict__ ig,
    const float* __restrict__ zg, const float* __restrict__ og,
    const float* __restrict__ rk,
    unsigned short* __restrict__ wbf, unsigned short* __restrict__ rkT) {
  const int WSZ = NHEAD * DHEAD * DHEAD;                      // 262144
  size_t i = (size_t)blockIdx.x * blockDim.x + threadIdx.x;   // 2M threads total
  if (i < (size_t)4 * WSZ) {
    int g = (int)(i / WSZ);
    int r = (int)(i % WSZ);
    const float* src = (g == 0) ? fg : (g == 1) ? ig : (g == 2) ? zg : og;
    wbf[i] = f2bf(src[r]);
  } else {
    size_t o = i - (size_t)4 * WSZ;                           // 0 .. 1M-1
    int h   = (int)(o / (4 * DHEAD * DHEAD));
    int rem = (int)(o % (4 * DHEAD * DHEAD));
    int j   = rem / DHEAD;                                    // 0..1023 (output col)
    int d   = rem % DHEAD;                                    // 0..255  (K)
    rkT[o] = f2bf(rk[(size_t)h * DHEAD * 4 * DHEAD + (size_t)d * 4 * DHEAD + j]);
  }
}

// ---------------------------------------------------------------------------
// Kernel 1: depthwise causal conv (K=4) + SiLU -> bf16 x_conv; also bf16(x).
// Pure bandwidth kernel: float4 loads, ushort4 stores.
// ---------------------------------------------------------------------------
__global__ __launch_bounds__(256) void conv_silu_kernel(
    const float* __restrict__ x, const float* __restrict__ cw,
    const float* __restrict__ cb,
    unsigned short* __restrict__ xconv, unsigned short* __restrict__ xbf) {
  size_t idx = (size_t)blockIdx.x * blockDim.x + threadIdx.x; // B*S*E/4 threads
  const int EV = EDIM / 4;
  int ev = (int)(idx % EV);
  int s  = (int)((idx / EV) % SDIM);
  int b  = (int)(idx / ((size_t)EV * SDIM));
  int e0 = ev * 4;
  size_t p = ((size_t)b * SDIM + s) * EDIM + e0;

  float4 acc = *(const float4*)(cb + e0);
  #pragma unroll
  for (int k = 0; k < KCONV; ++k) {
    int ss = s - (KCONV - 1) + k;
    if (ss >= 0) {
      float4 xv = *(const float4*)(x + ((size_t)b * SDIM + ss) * EDIM + e0);
      float4 wv = *(const float4*)(cw + (size_t)k * EDIM + e0);
      acc.x += xv.x * wv.x; acc.y += xv.y * wv.y;
      acc.z += xv.z * wv.z; acc.w += xv.w * wv.w;
    }
  }
  float4 xc = *(const float4*)(x + p);
  float sc[4] = {acc.x, acc.y, acc.z, acc.w};
  float xr[4] = {xc.x, xc.y, xc.z, xc.w};
  ushort4 oc, ox;
  unsigned short* ocp = (unsigned short*)&oc;
  unsigned short* oxp = (unsigned short*)&ox;
  #pragma unroll
  for (int q = 0; q < 4; ++q) {
    float v = sc[q];
    float silu = v * (1.0f / (1.0f + __expf(-v)));
    ocp[q] = f2bf(silu);
    oxp[q] = f2bf(xr[q]);
  }
  *(ushort4*)(xconv + p) = oc;
  *(ushort4*)(xbf + p)   = ox;
}

// ---------------------------------------------------------------------------
// Kernel 2: pre-gate GEMMs via bf16 WMMA. One wave per 16x16 output tile.
// All 8 waves of a block share one (b,s-tile,h,g) A panel -> stage it in LDS
// once with async b128 copies, read fragments back with ds_load_b128.
// gx layout: [s][b][h][g][d]  (scan-step contiguous).
// gi(g0)=x_conv*fgate, gf(g1)=x_conv*igate, gz(g2)=x*zgate, go(g3)=x*ogate.
// ---------------------------------------------------------------------------
__global__ __launch_bounds__(256) void pregate_kernel(
    const unsigned short* __restrict__ xconv,
    const unsigned short* __restrict__ xbf,
    const unsigned short* __restrict__ wbf,
    float* __restrict__ gx) {
  __shared__ unsigned short a_lds[16][DHEAD];   // 8 KB shared A panel (16 x 256 bf16)

  const int t    = threadIdx.x;
  const int lane = t & 31;
  const int wave = t >> 5;
  const int gw = blockIdx.x * 8 + wave;     // 262144 tiles total
  const int nt = gw & 15;                   // DH tile (N)
  const int g  = (gw >> 4) & 3;             // gate
  const int h  = (gw >> 6) & 3;             // head
  const int ms = gw >> 8;                   // 0..1023 = (b, s-tile)
  const int b  = ms >> 7;
  const int s0 = (ms & 127) << 4;

  const unsigned short* src = (g < 2) ? xconv : xbf;

  // cooperative async stage of the A panel: 512 x 16B units, 2 per thread
  #pragma unroll
  for (int q = 0; q < 2; ++q) {
    int u   = t * 2 + q;
    int row = u >> 5;                       // 32 units per 256-element row
    int seg = u & 31;
    const unsigned short* gp =
        src + ((size_t)(b * SDIM + s0 + row) * EDIM + h * DHEAD) + seg * 8;
    async_copy_b128(gp, &a_lds[row][seg * 8]);
  }
  wait_async0();
  __syncthreads();

  const int col = lane & 15;                // A row (M) and B col (N) index
  const int hl  = lane >> 4;                // half select
  const unsigned short* wrow =
      wbf + (size_t)((g * NHEAD + h) * DHEAD + nt * 16 + col) * DHEAD;

  v8f acc = {};
  #pragma unroll
  for (int kc = 0; kc < 8; ++kc) {
    const int kb = kc * 32 + hl * 8;
    FragCast fa;
    fa.q.lo = *(const uint4*)&a_lds[col][kb];        // ds_load_b128
    fa.q.hi = *(const uint4*)&a_lds[col][kb + 16];
    const unsigned short* pb = wrow + kc * 32 + hl * 8;
    FragCast fb; fb.q.lo = *(const uint4*)pb; fb.q.hi = *(const uint4*)(pb + 16);
    acc = __builtin_amdgcn_wmma_f32_16x16x32_bf16(false, fa.v, false, fb.v,
                                                  (short)0, acc, false, false);
  }
  // D fragment: lane (0-15) -> M=0..7 in v0..7; lane (16-31) -> M=8..15.
  const int row0 = hl * 8;
  size_t base = ((((size_t)(s0 + row0) * BDIM + b) * NHEAD + h) * NGATE + g) * DHEAD
                + nt * 16 + col;
  const size_t rstride = (size_t)BDIM * NHEAD * NGATE * DHEAD;  // next s
  #pragma unroll
  for (int i = 0; i < 8; ++i) gx[base + (size_t)i * rstride] = acc[i];
}

// ---------------------------------------------------------------------------
// Kernel 3: sequential sLSTM scan. One workgroup (32 waves) per head.
// y state in LDS (bf16, rows 8..15 zero), gates staged in LDS, cell states in
// registers. gx slice for step s+1 is async-prefetched into a double-buffered
// LDS region during the WMMA phase of step s (2 async b128 per wave per step;
// in-order completion => s_wait_asynccnt 2 guarantees current buffer ready).
// ---------------------------------------------------------------------------
static __device__ __forceinline__ float cell_update(
    float ir, float fr, float zr, float orr, float& c, float& n, float& m) {
  float ls = fminf(fr, 0.0f) - log1pf(__expf(-fabsf(fr)));  // log_sigmoid(fr)
  float lf = m + ls;
  float mn = fmaxf(ir, lf);
  float igv = __expf(ir - mn);
  float fgv = __expf(lf - mn);
  float cn = fgv * c + igv * tanhf(zr);
  float nn = fgv * n + igv;
  float ogv = 1.0f / (1.0f + __expf(-orr));
  float y = ogv * cn / nn;
  c = cn; n = nn; m = mn;
  return y;
}

__global__ __launch_bounds__(1024) void scan_kernel(
    const float* __restrict__ gx,
    const unsigned short* __restrict__ rkT,
    const float* __restrict__ rbias,
    float* __restrict__ ybuf) {
  __shared__ unsigned short y_lds[16][DHEAD];   //  8 KB (bf16 y, rows 8..15 = 0)
  __shared__ float raw_lds[8][4 * DHEAD];       // 32 KB (ry per batch row)
  __shared__ float gx_lds[2][BDIM * NGATE * DHEAD]; // 64 KB double-buffered gx slice

  const int h    = blockIdx.x;
  const int t    = threadIdx.x;
  const int lane = t & 31;
  const int wave = t >> 5;

  for (int i = t; i < 16 * DHEAD; i += 1024) ((unsigned short*)y_lds)[i] = 0;

  // each thread owns cell elements (bb0, d) and (bb0+4, d)
  const int bb0 = t >> 8;
  const int bb1 = bb0 + 4;
  const int d   = t & 255;
  float c0 = 0.f, c1 = 0.f, n0 = 0.f, n1 = 0.f, m0 = 0.f, m1 = 0.f;

  const float* rb = rbias + (size_t)h * NGATE * DHEAD;
  const float rb0 = rb[0 * DHEAD + d], rb1 = rb[1 * DHEAD + d];
  const float rb2 = rb[2 * DHEAD + d], rb3 = rb[3 * DHEAD + d];

  const int col = lane & 15;
  const int hl  = lane >> 4;
  const int nta = wave * 2, ntb = wave * 2 + 1;
  const unsigned short* browA = rkT + (size_t)(h * 4 * DHEAD + nta * 16 + col) * DHEAD;
  const unsigned short* browB = rkT + (size_t)(h * 4 * DHEAD + ntb * 16 + col) * DHEAD;

  const size_t sstride = (size_t)BDIM * NHEAD * NGATE * DHEAD;  // gx step stride

  // async-copy unit mapping for the per-step gx slice (2048 x 16B units)
  const int u0 = t, u1 = t + 1024;
  const int ub0 = u0 >> 8, uk0 = u0 & 255;    // batch chunk, 16B unit in chunk
  const int ub1 = u1 >> 8, uk1 = u1 & 255;
  const size_t gslot0 = ((size_t)(ub0 * NHEAD + h) * NGATE) * DHEAD + uk0 * 4;
  const size_t gslot1 = ((size_t)(ub1 * NHEAD + h) * NGATE) * DHEAD + uk1 * 4;

  // prologue: prefetch gx slice for step 0 into buffer 0
  async_copy_b128(gx + gslot0, &gx_lds[0][u0 * 4]);
  async_copy_b128(gx + gslot1, &gx_lds[0][u1 * 4]);

  __syncthreads();

  for (int s = 0; s < SDIM; ++s) {
    // ---- issue async prefetch of next step's gx slice ----
    {
      const int sp  = (s + 1 < SDIM) ? s + 1 : 0;   // dummy re-issue on last step
      const int nb  = (s + 1) & 1;
      const size_t gbase = (size_t)sp * sstride;
      async_copy_b128(gx + gbase + gslot0, &gx_lds[nb][u0 * 4]);
      async_copy_b128(gx + gbase + gslot1, &gx_lds[nb][u1 * 4]);
    }

    // ---- ry = y x rkernel[h]  (16x1024 over K=256), 2 N-tiles per wave ----
    v8f acc0 = {}, acc1 = {};
    #pragma unroll
    for (int kc = 0; kc < 8; ++kc) {
      const int kb = kc * 32 + hl * 8;
      FragCast fa;
      fa.q.lo = *(const uint4*)&y_lds[col][kb];        // ds_load_b128
      fa.q.hi = *(const uint4*)&y_lds[col][kb + 16];
      const unsigned short* pb0 = browA + kc * 32 + hl * 8;
      FragCast fb0; fb0.q.lo = *(const uint4*)pb0; fb0.q.hi = *(const uint4*)(pb0 + 16);
      const unsigned short* pb1 = browB + kc * 32 + hl * 8;
      FragCast fb1; fb1.q.lo = *(const uint4*)pb1; fb1.q.hi = *(const uint4*)(pb1 + 16);
      acc0 = __builtin_amdgcn_wmma_f32_16x16x32_bf16(false, fa.v, false, fb0.v,
                                                     (short)0, acc0, false, false);
      acc1 = __builtin_amdgcn_wmma_f32_16x16x32_bf16(false, fa.v, false, fb1.v,
                                                     (short)0, acc1, false, false);
    }
    // lanes 0-15 hold M=0..7 (real batches) in v0..7 -> stage gates to LDS
    if (lane < 16) {
      #pragma unroll
      for (int i = 0; i < 8; ++i) {
        raw_lds[i][nta * 16 + col] = acc0[i];
        raw_lds[i][ntb * 16 + col] = acc1[i];
      }
    }
    // drain the prefetch that fills the CURRENT buffer (2 newest may remain)
    wait_async2();
    __syncthreads();

    // ---- cell update: 2 (b,d) elements per thread, gx read from LDS ----
    const float* gxc = &gx_lds[s & 1][0];
    {
      const float* gc = gxc + bb0 * (NGATE * DHEAD);
      float ir  = gc[d]             + raw_lds[bb0][0 * DHEAD + d] + rb0;
      float fr  = gc[DHEAD + d]     + raw_lds[bb0][1 * DHEAD + d] + rb1;
      float zr  = gc[2 * DHEAD + d] + raw_lds[bb0][2 * DHEAD + d] + rb2;
      float orr = gc[3 * DHEAD + d] + raw_lds[bb0][3 * DHEAD + d] + rb3;
      float y = cell_update(ir, fr, zr, orr, c0, n0, m0);
      ybuf[(((size_t)bb0 * SDIM + s) * NHEAD + h) * DHEAD + d] = y;
      y_lds[bb0][d] = f2bf(y);
    }
    {
      const float* gc = gxc + bb1 * (NGATE * DHEAD);
      float ir  = gc[d]             + raw_lds[bb1][0 * DHEAD + d] + rb0;
      float fr  = gc[DHEAD + d]     + raw_lds[bb1][1 * DHEAD + d] + rb1;
      float zr  = gc[2 * DHEAD + d] + raw_lds[bb1][2 * DHEAD + d] + rb2;
      float orr = gc[3 * DHEAD + d] + raw_lds[bb1][3 * DHEAD + d] + rb3;
      float y = cell_update(ir, fr, zr, orr, c1, n1, m1);
      ybuf[(((size_t)bb1 * SDIM + s) * NHEAD + h) * DHEAD + d] = y;
      y_lds[bb1][d] = f2bf(y);
    }
    __syncthreads();
  }
}

// ---------------------------------------------------------------------------
// Kernel 4: per-(b,s,h) GroupNorm over DH=256 + gn_weight scale.
// ---------------------------------------------------------------------------
__global__ __launch_bounds__(256) void gnorm_kernel(
    const float* __restrict__ ybuf, const float* __restrict__ gnw,
    float* __restrict__ out) {
  __shared__ float r0[256];
  __shared__ float r1[256];
  const int gid = blockIdx.x;          // (b*S+s)*NH + h
  const int h   = gid & 3;
  const int bs  = gid >> 2;
  const int d   = threadIdx.x;
  float v = ybuf[(size_t)gid * DHEAD + d];
  r0[d] = v; r1[d] = v * v;
  __syncthreads();
  #pragma unroll
  for (int off = 128; off > 0; off >>= 1) {
    if (d < off) { r0[d] += r0[d + off]; r1[d] += r1[d + off]; }
    __syncthreads();
  }
  float mu  = r0[0] * (1.0f / DHEAD);
  float var = r1[0] * (1.0f / DHEAD) - mu * mu;
  float inv = rsqrtf(var + 1e-5f);
  out[(size_t)bs * EDIM + h * DHEAD + d] = (v - mu) * inv * gnw[h * DHEAD + d];
}

// ---------------------------------------------------------------------------
extern "C" void kernel_launch(void* const* d_in, const int* in_sizes, int n_in,
                              void* d_out, int out_size, void* d_ws, size_t ws_size,
                              hipStream_t stream) {
  const float* x      = (const float*)d_in[0];
  const float* conv_w = (const float*)d_in[1];
  const float* conv_b = (const float*)d_in[2];
  const float* fgate  = (const float*)d_in[3];
  const float* igate  = (const float*)d_in[4];
  const float* zgate  = (const float*)d_in[5];
  const float* ogate  = (const float*)d_in[6];
  const float* rk     = (const float*)d_in[7];
  const float* rbias  = (const float*)d_in[8];
  const float* gnw    = (const float*)d_in[9];
  float* out = (float*)d_out;
  char* ws = (char*)d_ws;

  // workspace layout (bytes)
  unsigned short* xconv = (unsigned short*)(ws + 0);                    // 32 MB
  unsigned short* xbf   = (unsigned short*)(ws + (size_t)33554432);     // 32 MB
  unsigned short* wbf   = (unsigned short*)(ws + (size_t)67108864);     // 2 MB
  unsigned short* rkT   = (unsigned short*)(ws + (size_t)69206016);     // 2 MB
  float*          gx    = (float*)(ws + (size_t)71303168);              // 256 MB
  float*          ybuf  = (float*)(ws + (size_t)339738624);             // 64 MB

  prep_kernel<<<8192, 256, 0, stream>>>(fgate, igate, zgate, ogate, rk, wbf, rkT);
  conv_silu_kernel<<<16384, 256, 0, stream>>>(x, conv_w, conv_b, xconv, xbf);
  pregate_kernel<<<32768, 256, 0, stream>>>(xconv, xbf, wbf, gx);
  scan_kernel<<<NHEAD, 1024, 0, stream>>>(gx, rkT, rbias, ybuf);
  gnorm_kernel<<<BDIM * SDIM * NHEAD, 256, 0, stream>>>(ybuf, gnw, out);
}
